// SNNTokenizer_16887811408309
// MI455X (gfx1250) — matrix operand order repaired and checked by reading, not compiled, not run on hardware
//
#include <hip/hip_runtime.h>
#include <hip/hip_bf16.h>
#include <math.h>

// ---------------------------------------------------------------------------
// SNN tokenizer forward for MI455X (gfx1250, wave32).
// GEMMs use v_wmma_f32_16x16x32_f16 (f16 A/B, f32 acc): workload is matrix-op
// bound (~61 GF vs ~150 MB HBM traffic => ~7us memory floor at 23.3 TB/s).
// Weights are pre-swizzled into WMMA fragment-major layout so every B
// fragment is one contiguous 32-byte read per lane (2x b128), coalesced
// across the wave. The big MLP GEMM is M=32 register-blocked with the H
// dimension processed in 4 chunks of 256 (z1 chunk lives in 16 KB of LDS),
// halving W2 L2 traffic vs an M=16 tiling and reusing every A fragment x6
// and every B fragment x2 from registers.
// Elementwise math (LN, exact gelu, LIF scan, segment pooling) stays fp32.
// ---------------------------------------------------------------------------

typedef __attribute__((ext_vector_type(16))) _Float16 v16h;
typedef __attribute__((ext_vector_type(8)))  float    v8f;

#define B_ 16
#define T_ 2048
#define D_ 128
#define H_ 1024
#define O_ 768
#define K_ 2048   // segment upper bound == T
#define HC_ 256   // H chunk width for proj kernel (H_/4)

__device__ __forceinline__ float gelu_exact(float x) {
  return 0.5f * x * (1.0f + erff(x * 0.7071067811865475f));
}

// A fragment: 16x32 (MxK) f16 tile, row-major with stride ld.
// ISA 7.12.2: lanes 0-15 hold M=lane, K in {0..7,16..23}; lanes 16-31 hold
// K in {8..15,24..31}. Both 8-element groups contiguous => 2x ds_load_b128.
__device__ __forceinline__ v16h load_a16x32(const _Float16* base, int ld) {
  const int lane = threadIdx.x & 31;
  const int m = lane & 15;
  const int kb = (lane & 16) ? 8 : 0;
  v16h a;
#pragma unroll
  for (int j = 0; j < 16; ++j) {
    const int k = kb + ((j < 8) ? j : (j + 8));
    a[j] = base[m * ld + k];
  }
  return a;
}

// B fragment from pre-swizzled fragment-major storage: fragment `fragidx`
// holds 32 lanes x 16 halves contiguously; lane reads one aligned 32B chunk.
__device__ __forceinline__ v16h load_b_frag(const _Float16* frags, int fragidx) {
  const int lane = threadIdx.x & 31;
  return *(const v16h*)(frags + (((size_t)fragidx * 32 + lane) << 4));
}

// ---------------------------------------------------------------------------
// Kernel 0: swizzle a row-major KxN f32 weight matrix into WMMA B-fragment-
// major f16 layout. Fragment grid: (ntile = n/16) x (kt = k/32); within a
// fragment: lane l holds N = ntile*16 + (l&15), K = kt*32 + 8*(l>=16) + (j<8 ? j : j+8).
// ---------------------------------------------------------------------------
__global__ __launch_bounds__(256)
void swizzle_kernel(const float* __restrict__ W, _Float16* __restrict__ out,
                    int Kdim, int Ndim) {
  const int nkt = Kdim / 32;
  const size_t total = (size_t)Kdim * Ndim;
  for (size_t i = (size_t)blockIdx.x * blockDim.x + threadIdx.x; i < total;
       i += (size_t)gridDim.x * blockDim.x) {
    const int fragidx = (int)(i >> 9);   // 512 halves per fragment
    const int within  = (int)(i & 511);
    const int lane = within >> 4;
    const int j    = within & 15;
    const int ntile = fragidx / nkt;
    const int kt    = fragidx % nkt;
    const int n = ntile * 16 + (lane & 15);
    const int k = kt * 32 + ((lane & 16) ? 8 : 0) + ((j < 8) ? j : (j + 8));
    out[i] = (_Float16)W[(size_t)k * Ndim + n];
  }
}

// ---------------------------------------------------------------------------
// Kernel 1: embedding gather (padding_idx=0) -> 128x128 GEMM (WMMA) -> +bias
// -> LayerNorm -> exact gelu -> h; plus boundary/reset logit dot products.
// One 16-row tile per block; 256 threads = 8 waves; wave w owns N-tile w*16.
// ---------------------------------------------------------------------------
__global__ __launch_bounds__(256)
void ctx_kernel(const int* __restrict__ inputs,
                const float* __restrict__ emb,
                const _Float16* __restrict__ Wsw,   // swizzled ctx_W (32 frags)
                const float* __restrict__ ctx_b,
                const float* __restrict__ ln_g,
                const float* __restrict__ ln_b,
                const float* __restrict__ bnd_W, const float* __restrict__ bnd_b,
                const float* __restrict__ rst_W, const float* __restrict__ rst_b,
                float* __restrict__ h_out,
                float* __restrict__ bnd_out,
                float* __restrict__ rst_out) {
  __shared__ __align__(32) _Float16 sW[D_ * D_];   // 32 KB swizzled ctx_W
  __shared__ __align__(32) _Float16 sA[16 * D_];   // 4 KB embedded rows (f16)
  __shared__ float                  sY[16 * D_];   // 8 KB GEMM result (f32)

  const int tid  = threadIdx.x;
  const int row0 = blockIdx.x * 16;

  for (int i = tid; i < D_ * D_; i += 256) sW[i] = Wsw[i];
  for (int i = tid; i < 16 * D_; i += 256) {
    const int m = i >> 7, d = i & 127;
    const int tok = inputs[row0 + m];
    const float v = (tok == 0) ? 0.0f : emb[tok * D_ + d];  // padding row = 0
    sA[i] = (_Float16)v;
  }
  __syncthreads();

  const int wid = tid >> 5;
  {
    const int n0 = wid * 16;
    v8f acc = {};
#pragma unroll
    for (int kk = 0; kk < 4; ++kk) {
      v16h a = load_a16x32(sA + kk * 32, D_);
      v16h b = load_b_frag(sW, wid * 4 + kk);   // nkt = 4 for 128x128
      acc = __builtin_amdgcn_wmma_f32_16x16x32_f16(false, a, false, b,
                                                   (short)0, acc, false, false);
    }
    const int lane = tid & 31;
    const int n = lane & 15;
    const int mb = (lane & 16) ? 8 : 0;
#pragma unroll
    for (int r = 0; r < 8; ++r) sY[(mb + r) * D_ + n0 + n] = acc[r];
  }
  __syncthreads();

  // LN + gelu + logits. Wave handles 2 rows; 16 lanes/row, 8 cols/lane.
  const int lane = tid & 31;
  const int half = lane >> 4;
  const int l    = lane & 15;
  const int row  = wid * 2 + half;
  const int c0   = l * 8;
  float y[8];
  float s = 0.f, s2 = 0.f;
#pragma unroll
  for (int j = 0; j < 8; ++j) {
    const float v = sY[row * D_ + c0 + j] + ctx_b[c0 + j];
    y[j] = v; s += v; s2 += v * v;
  }
#pragma unroll
  for (int m = 1; m <= 8; m <<= 1) {
    s  += __shfl_xor(s,  m, 16);
    s2 += __shfl_xor(s2, m, 16);
  }
  const float mean = s * (1.0f / D_);
  const float var  = s2 * (1.0f / D_) - mean * mean;
  const float inv  = rsqrtf(var + 1e-5f);
  const int g = row0 + row;
  float db = 0.f, dr = 0.f;
#pragma unroll
  for (int j = 0; j < 8; ++j) {
    const int c = c0 + j;
    const float t  = (y[j] - mean) * inv * ln_g[c] + ln_b[c];
    const float hv = gelu_exact(t);
    h_out[(size_t)g * D_ + c] = hv;
    db += hv * bnd_W[c];
    dr += hv * rst_W[c];
  }
#pragma unroll
  for (int m = 1; m <= 8; m <<= 1) {
    db += __shfl_xor(db, m, 16);
    dr += __shfl_xor(dr, m, 16);
  }
  if (l == 0) {
    bnd_out[g] = tanhf(db + bnd_b[0]);
    rst_out[g] = dr + rst_b[0];
  }
}

// ---------------------------------------------------------------------------
// Kernel 2: conditional-LIF scan over T (sequential), lane b = batch row.
// ---------------------------------------------------------------------------
__global__ void lif_kernel(const float* __restrict__ bnd,
                           const float* __restrict__ rst,
                           float* __restrict__ wout,
                           int* __restrict__ ids) {
  const int b = threadIdx.x;
  if (b >= B_) return;
  float v = 0.f;
  int cnt = 0;
  for (int t = 0; t < T_; ++t) {
    const float x = bnd[b * T_ + t];
    const bool r = rst[b * T_ + t] > 0.0f;      // sigmoid(z)>0.5 <=> z>0
    v += (x - v) * 0.5f;                        // charge, TAU=2
    const bool sp = (v >= 1.0f);                // VTH=1
    ids[b * T_ + t]  = cnt;                     // rolled cumsum (ids before t)
    wout[b * T_ + t] = (v > 0.f) ? v : 0.f;     // relu(pre-reset v)
    if (sp) ++cnt;
    if (sp && r) v = 0.f;                       // conditional hard reset
  }
}

// ---------------------------------------------------------------------------
// Kernel 3: segment pooling (ids monotone per batch).
// ---------------------------------------------------------------------------
__global__ __launch_bounds__(128)
void pool_kernel(const float* __restrict__ h, const float* __restrict__ w,
                 const int* __restrict__ ids, float* __restrict__ tok) {
  const int b = blockIdx.x;
  const int d = threadIdx.x;
  float* tb = tok + (size_t)b * K_ * D_;
  for (int i = d; i < K_ * D_; i += 128) tb[i] = 0.0f;  // zero own column
  int cur = 0;
  float accv = 0.f, accw = 0.f;
  for (int t = 0; t < T_; ++t) {
    const int   id = ids[b * T_ + t];
    const float wv = w[b * T_ + t];
    if (id != cur) {
      tb[cur * D_ + d] = accv / (accw + 1e-8f);
      accv = 0.f; accw = 0.f; cur = id;
    }
    accv += wv * h[((size_t)b * T_ + t) * D_ + d];
    accw += wv;
  }
  tb[cur * D_ + d] = accv / (accw + 1e-8f);
}

// ---------------------------------------------------------------------------
// Kernel 4: LN1 -> (128x1024 WMMA) -> gelu -> (1024x768 WMMA) + b2, writing
// pre-LN z2 to d_out. M=32 rows per block, 8 waves, H chunked by 256.
// Per K-step of mm2: 2 A-loads + 6 B-loads feed 12 WMMAs (A reused x6 from
// registers across N-tiles, B reused x2 across M-halves).
// ---------------------------------------------------------------------------
__global__ __launch_bounds__(256)
void proj_kernel(const float* __restrict__ tok,
                 const float* __restrict__ ln1_g, const float* __restrict__ ln1_b,
                 const _Float16* __restrict__ W1sw, const float* __restrict__ b1,
                 const _Float16* __restrict__ W2sw, const float* __restrict__ b2,
                 float* __restrict__ out) {
  __shared__ __align__(32) _Float16 sA[32 * D_];     // 8 KB LN'd rows (f16)
  __shared__ __align__(32) _Float16 sZ1[32 * HC_];   // 16 KB z1 chunk (f16)

  const int tid  = threadIdx.x;
  const int row0 = blockIdx.x * 32;
  const int wid  = tid >> 5;
  const int lane = tid & 31;
  const int half = lane >> 4;
  const int l    = lane & 15;
  const int n    = lane & 15;
  const int mb   = (lane & 16) ? 8 : 0;

  // ---- LayerNorm over D=128 -> f16 A tile (4 rows per wave, 2 passes) ----
#pragma unroll
  for (int rp = 0; rp < 2; ++rp) {
    const int row = rp * 16 + wid * 2 + half;
    const int c0  = l * 8;
    const float* src = tok + (size_t)(row0 + row) * D_ + c0;
    float y[8]; float s = 0.f, s2 = 0.f;
#pragma unroll
    for (int j = 0; j < 8; ++j) {
      const float v = src[j];
      y[j] = v; s += v; s2 += v * v;
    }
#pragma unroll
    for (int m = 1; m <= 8; m <<= 1) {
      s  += __shfl_xor(s,  m, 16);
      s2 += __shfl_xor(s2, m, 16);
    }
    const float mean = s * (1.0f / D_);
    const float var  = s2 * (1.0f / D_) - mean * mean;
    const float inv  = rsqrtf(var + 1e-5f);
#pragma unroll
    for (int j = 0; j < 8; ++j) {
      const int c = c0 + j;
      sA[row * D_ + c] = (_Float16)((y[j] - mean) * inv * ln1_g[c] + ln1_b[c]);
    }
  }
  __syncthreads();

  // ---- persistent mm2 accumulators: 6 N-tiles x 2 M-halves ----
  v8f acc[6][2];
#pragma unroll
  for (int i = 0; i < 6; ++i) {
    acc[i][0] = (v8f){};
    acc[i][1] = (v8f){};
  }

#pragma unroll 1
  for (int c = 0; c < 4; ++c) {          // H chunks of 256
    // mm1: z1 chunk = gelu(A @ W1[:, c*256 : c*256+256] + b1)
#pragma unroll 1
    for (int nt = 0; nt < 2; ++nt) {
      const int ntile_g = c * 16 + wid * 2 + nt;   // H-tile index (0..63)
      v8f a1[2] = {(v8f){}, (v8f){}};
#pragma unroll
      for (int kk = 0; kk < 4; ++kk) {
        v16h b  = load_b_frag(W1sw, ntile_g * 4 + kk);   // nkt = 4
        v16h a0 = load_a16x32(sA + kk * 32, D_);          // rows 0-15
        v16h a1f = load_a16x32(sA + 16 * D_ + kk * 32, D_);// rows 16-31
        a1[0] = __builtin_amdgcn_wmma_f32_16x16x32_f16(false, a0, false, b,
                                                       (short)0, a1[0], false, false);
        a1[1] = __builtin_amdgcn_wmma_f32_16x16x32_f16(false, a1f, false, b,
                                                       (short)0, a1[1], false, false);
      }
      const int col0 = (wid * 2 + nt) * 16 + n;   // column within chunk
#pragma unroll
      for (int mt = 0; mt < 2; ++mt) {
#pragma unroll
        for (int r = 0; r < 8; ++r) {
          const int m = mt * 16 + mb + r;
          sZ1[m * HC_ + col0] =
              (_Float16)gelu_exact(a1[mt][r] + b1[c * HC_ + col0]);
        }
      }
    }
    __syncthreads();

    // mm2 partial: acc += z1chunk @ W2[c*256 : c*256+256, :]
#pragma unroll 1
    for (int kk2 = 0; kk2 < 8; ++kk2) {
      const int kt = c * 8 + kk2;                  // global K-tile (0..31)
      v16h a0 = load_a16x32(sZ1 + kk2 * 32, HC_);           // rows 0-15
      v16h a1 = load_a16x32(sZ1 + 16 * HC_ + kk2 * 32, HC_);// rows 16-31
#pragma unroll
      for (int i = 0; i < 6; ++i) {
        v16h b = load_b_frag(W2sw, (wid * 6 + i) * 32 + kt);  // nkt = 32
        acc[i][0] = __builtin_amdgcn_wmma_f32_16x16x32_f16(false, a0, false, b,
                                                           (short)0, acc[i][0], false, false);
        acc[i][1] = __builtin_amdgcn_wmma_f32_16x16x32_f16(false, a1, false, b,
                                                           (short)0, acc[i][1], false, false);
      }
    }
    __syncthreads();   // before next chunk overwrites sZ1
  }

  // ---- epilogue: +b2, store pre-LN z2 to d_out ----
#pragma unroll
  for (int i = 0; i < 6; ++i) {
    const int n0 = (wid * 6 + i) * 16;
#pragma unroll
    for (int mt = 0; mt < 2; ++mt) {
#pragma unroll
      for (int r = 0; r < 8; ++r) {
        const int m = mt * 16 + mb + r;
        const int cc = n0 + n;
        out[(size_t)(row0 + m) * O_ + cc] = acc[i][mt][r] + b2[cc];
      }
    }
  }
}

// ---------------------------------------------------------------------------
// Kernel 5: in-place LayerNorm over the last dim (768) of d_out.
// One wave per row (32 lanes x 24 cols), 8 rows per 256-thread block.
// ---------------------------------------------------------------------------
__global__ __launch_bounds__(256)
void ln_out_kernel(float* __restrict__ out,
                   const float* __restrict__ g2, const float* __restrict__ b2) {
  const int wid  = threadIdx.x >> 5;
  const int lane = threadIdx.x & 31;
  const size_t row = (size_t)blockIdx.x * 8 + wid;
  float* p = out + row * O_;
  float x[24]; float s = 0.f, s2 = 0.f;
#pragma unroll
  for (int j = 0; j < 24; ++j) {
    const float v = p[lane + j * 32];
    x[j] = v; s += v; s2 += v * v;
  }
#pragma unroll
  for (int m = 1; m <= 16; m <<= 1) {
    s  += __shfl_xor(s,  m, 32);
    s2 += __shfl_xor(s2, m, 32);
  }
  const float mean = s * (1.0f / O_);
  const float var  = s2 * (1.0f / O_) - mean * mean;
  const float inv  = rsqrtf(var + 1e-5f);
#pragma unroll
  for (int j = 0; j < 24; ++j) {
    const int c = lane + j * 32;
    p[c] = (x[j] - mean) * inv * g2[c] + b2[c];
  }
}

// ---------------------------------------------------------------------------
extern "C" void kernel_launch(void* const* d_in, const int* in_sizes, int n_in,
                              void* d_out, int out_size, void* d_ws, size_t ws_size,
                              hipStream_t stream) {
  (void)in_sizes; (void)n_in; (void)out_size; (void)ws_size;

  const int*   inputs = (const int*)d_in[0];
  const float* emb    = (const float*)d_in[1];
  const float* ctxW   = (const float*)d_in[2];
  const float* ctxb   = (const float*)d_in[3];
  const float* lng    = (const float*)d_in[4];
  const float* lnb    = (const float*)d_in[5];
  const float* bndW   = (const float*)d_in[6];
  const float* bndb   = (const float*)d_in[7];
  const float* rstW   = (const float*)d_in[8];
  const float* rstb   = (const float*)d_in[9];
  const float* ln1g   = (const float*)d_in[10];
  const float* ln1b   = (const float*)d_in[11];
  const float* W1     = (const float*)d_in[12];
  const float* b1     = (const float*)d_in[13];
  const float* W2     = (const float*)d_in[14];
  const float* b2     = (const float*)d_in[15];
  const float* ln2g   = (const float*)d_in[16];
  const float* ln2b   = (const float*)d_in[17];
  float* out = (float*)d_out;

  // ---- workspace carve-up (~35 MB; everything re-initialized every call) ----
  char* ws = (char*)d_ws;
  size_t off = 0;
  auto carve = [&](size_t bytes) -> void* {
    void* p = ws + off;
    off = (off + bytes + 255) & ~(size_t)255;
    return p;
  };
  _Float16* ctxWsw = (_Float16*)carve((size_t)D_ * D_ * 2);
  _Float16* W1sw   = (_Float16*)carve((size_t)D_ * H_ * 2);
  _Float16* W2sw   = (_Float16*)carve((size_t)H_ * O_ * 2);
  float* h_buf     = (float*)carve((size_t)B_ * T_ * D_ * 4);
  float* bnd_buf   = (float*)carve((size_t)B_ * T_ * 4);
  float* rst_buf   = (float*)carve((size_t)B_ * T_ * 4);
  float* w_buf     = (float*)carve((size_t)B_ * T_ * 4);
  int*   ids_buf   = (int*)carve((size_t)B_ * T_ * 4);
  float* tok_buf   = (float*)carve((size_t)B_ * K_ * D_ * 4);

  // 0) weights -> f16, WMMA fragment-major swizzle
  swizzle_kernel<<<64,   256, 0, stream>>>(ctxW, ctxWsw, D_, D_);
  swizzle_kernel<<<512,  256, 0, stream>>>(W1, W1sw, D_, H_);
  swizzle_kernel<<<3072, 256, 0, stream>>>(W2, W2sw, H_, O_);

  // 1) embed + ctx GEMM + LN + gelu + logits   (B*T/16 = 2048 tiles)
  ctx_kernel<<<(B_ * T_) / 16, 256, 0, stream>>>(
      inputs, emb, ctxWsw, ctxb, lng, lnb, bndW, bndb, rstW, rstb,
      h_buf, bnd_buf, rst_buf);

  // 2) LIF scan (sequential over T; one lane per batch row)
  lif_kernel<<<1, 32, 0, stream>>>(bnd_buf, rst_buf, w_buf, ids_buf);

  // 3) segment pooling
  pool_kernel<<<B_, 128, 0, stream>>>(h_buf, w_buf, ids_buf, tok_buf);

  // 4) LN1 + MLP GEMMs -> pre-LN z2 in d_out   (M=32 per block)
  proj_kernel<<<(B_ * K_) / 32, 256, 0, stream>>>(
      tok_buf, ln1g, ln1b, W1sw, b1, W2sw, b2, out);

  // 5) final LayerNorm over 768, in place on d_out
  ln_out_kernel<<<(B_ * K_) / 8, 256, 0, stream>>>(out, ln2g, ln2b);
}
